// GAT_17540646436882
// MI455X (gfx1250) — matrix operand order repaired and checked by reading, not compiled, not run on hardware
//
#include <hip/hip_runtime.h>
#include <hip/hip_bf16.h>
#include <stdint.h>

// ---------------------------------------------------------------------------
// GAT (3-layer, PyG GATConv) for MI455X / gfx1250.
//  - GEMMs: bf16 WMMA (v_wmma_f32_16x16x32_bf16), fp32 accumulate.
//    LDS tiles staged so each fragment = two ds_load_b128 (contiguous k per lane group).
//  - Softmax/aggregation: edge-parallel, float4 gathers + float atomics,
//    32-bit shift/mask index math (all divisors are powers of two).
// ---------------------------------------------------------------------------

typedef __attribute__((ext_vector_type(16))) __bf16 v16bf;
typedef __attribute__((ext_vector_type(8)))  __bf16 v8bf;
typedef __attribute__((ext_vector_type(8)))  float  v8f;

#define NNODES   50000
#define EREAL    800000
#define EEXT     850000            // + self loops
#define NEGSLOPE 0.2f

// ---------------------------- WMMA GEMM ------------------------------------
// C[M,N] = A[M,K] * B[K,N], fp32 in/out, bf16 WMMA math.
// Block: 256 threads = 8 waves; macro-tile 128x32; wave tile 16x32 (2 WMMAs).
__global__ __launch_bounds__(256) void wmma_gemm_bf16(
    const float* __restrict__ A, const float* __restrict__ B,
    float* __restrict__ C, int M, int N, int K)
{
    __shared__ __bf16 sA[128][32];   // [m][k] row-major (64B rows)
    __shared__ __bf16 sBt[32][32];   // [n][k] transposed (64B rows)

    const int t    = threadIdx.x;
    const int lane = t & 31;
    const int wave = t >> 5;              // wave row 0..7 (16 M-rows each)
    const int g    = lane >> 4;           // lane group 0/1
    const int l15  = lane & 15;
    const int rowBase = blockIdx.x * 128;
    const int colBase = blockIdx.y * 32;

    v8f acc0 = {};
    v8f acc1 = {};

    for (int k0 = 0; k0 < K; k0 += 32) {
        // ---- Stage A: 128x32 fp32 -> bf16. Thread t: row t>>1, 16 cols. ----
        {
            int r = t >> 1;
            int c = (t & 1) << 4;
            int row = rowBase + r;
            if (row >= M) row = M - 1;            // clamp: no divergent loads
            const float4* s4 = (const float4*)(A + (size_t)row * K + (k0 + c));
            float4 f0 = s4[0], f1 = s4[1], f2 = s4[2], f3 = s4[3];
            union { v8bf v; __bf16 e[8]; } p0, p1;
            p0.e[0] = (__bf16)f0.x; p0.e[1] = (__bf16)f0.y;
            p0.e[2] = (__bf16)f0.z; p0.e[3] = (__bf16)f0.w;
            p0.e[4] = (__bf16)f1.x; p0.e[5] = (__bf16)f1.y;
            p0.e[6] = (__bf16)f1.z; p0.e[7] = (__bf16)f1.w;
            p1.e[0] = (__bf16)f2.x; p1.e[1] = (__bf16)f2.y;
            p1.e[2] = (__bf16)f2.z; p1.e[3] = (__bf16)f2.w;
            p1.e[4] = (__bf16)f3.x; p1.e[5] = (__bf16)f3.y;
            p1.e[6] = (__bf16)f3.z; p1.e[7] = (__bf16)f3.w;
            *(v8bf*)&sA[r][c]     = p0.v;
            *(v8bf*)&sA[r][c + 8] = p1.v;
        }
        // ---- Stage B transposed: sBt[n][k]. Thread t: k row t>>3, 4 n. ----
        {
            int r  = t >> 3;                       // k 0..31
            int c4 = (t & 7) << 2;                 // n 0..28
            float4 f = *(const float4*)(B + (size_t)(k0 + r) * N + (colBase + c4));
            sBt[c4 + 0][r] = (__bf16)f.x;
            sBt[c4 + 1][r] = (__bf16)f.y;
            sBt[c4 + 2][r] = (__bf16)f.z;
            sBt[c4 + 3][r] = (__bf16)f.w;
        }
        __syncthreads();

        // ---- Fragments: two ds_load_b128 each (contiguous k per lane group). ----
        // A 16x32: lane group g holds K=16g..16g+15 (contiguous k order).
        union { v16bf v; v8bf h[2]; } fa, fb0, fb1;
        const v8bf* rowA = (const v8bf*)&sA[(wave << 4) + l15][0];
        fa.h[0] = rowA[2 * g];
        fa.h[1] = rowA[2 * g + 1];
        // B 32x16: N = l15 (+16 for second tile); lane group g holds K=16g..16g+15 (contiguous, same as A).
        const v8bf* rowB0 = (const v8bf*)&sBt[l15][0];
        fb0.h[0] = rowB0[2 * g];
        fb0.h[1] = rowB0[2 * g + 1];
        const v8bf* rowB1 = (const v8bf*)&sBt[16 + l15][0];
        fb1.h[0] = rowB1[2 * g];
        fb1.h[1] = rowB1[2 * g + 1];

        acc0 = __builtin_amdgcn_wmma_f32_16x16x32_bf16(
                   false, fa.v, false, fb0.v, (short)0, acc0, false, false);
        acc1 = __builtin_amdgcn_wmma_f32_16x16x32_bf16(
                   false, fa.v, false, fb1.v, (short)0, acc1, false, false);
        __syncthreads();
    }

    // C/D layout: VGPR r -> M = r (+8 for lanes 16-31), N = lane&15.
    // Base pointer once, step by N per row; uniform fast path for full blocks.
    const int row0 = rowBase + (wave << 4) + (g << 3);
    float* cp = C + (size_t)row0 * N + colBase + l15;
    if (rowBase + 128 <= M) {            // wave-uniform: no exec-mask dance
        #pragma unroll
        for (int r = 0; r < 8; ++r) {
            cp[0]  = acc0[r];
            cp[16] = acc1[r];
            cp += N;
        }
    } else {                              // tail block only
        #pragma unroll
        for (int r = 0; r < 8; ++r) {
            if (row0 + r < M) {
                cp[0]  = acc0[r];
                cp[16] = acc1[r];
            }
            cp += N;
        }
    }
}

// ------------------------ attention coefficients ----------------------------
__global__ void attn_coef_kernel(const float* __restrict__ h,
                                 const float* __restrict__ a_s,
                                 const float* __restrict__ a_d,
                                 float* __restrict__ al_s,
                                 float* __restrict__ al_d,
                                 int hsh, int csh)        // Hh = 1<<hsh, Cc = 1<<csh
{
    unsigned idx = blockIdx.x * blockDim.x + threadIdx.x;  // node*H + head
    if (idx >= (unsigned)(NNODES << hsh)) return;
    unsigned node = idx >> hsh, hd = idx & ((1u << hsh) - 1);
    const float4* hp = (const float4*)(h + ((size_t)node << (hsh + csh)) + (hd << csh));
    const float4* sp = (const float4*)(a_s + (hd << csh));
    const float4* dp = (const float4*)(a_d + (hd << csh));
    float ss = 0.f, dd = 0.f;
    int c4 = 1 << (csh - 2);
    for (int c = 0; c < c4; ++c) {
        float4 hv = hp[c], sv = sp[c], dv = dp[c];
        ss += hv.x * sv.x + hv.y * sv.y + hv.z * sv.z + hv.w * sv.w;
        dd += hv.x * dv.x + hv.y * dv.y + hv.z * dv.z + hv.w * dv.w;
    }
    al_s[idx] = ss;
    al_d[idx] = dd;
}

// ------------------------- segment softmax helpers --------------------------
__global__ void init_md_kernel(float* __restrict__ m, float* __restrict__ denom, int n)
{
    int i = blockIdx.x * blockDim.x + threadIdx.x;
    if (i < n) { m[i] = __int_as_float((int)0xff800000); denom[i] = 0.f; }  // -inf, 0
}

__device__ __forceinline__ void atomicMaxF(float* a, float v)
{
    // IEEE-order atomic max via int punning (handles mixed signs).
    if (v >= 0.f) atomicMax((int*)a, __float_as_int(v));
    else          atomicMin((unsigned int*)a, __float_as_uint(v));
}

__device__ __forceinline__ void edge_sd(int e, const int* __restrict__ src,
                                        const int* __restrict__ dst, int& s, int& d)
{
    if (e < EREAL) { s = src[e]; d = dst[e]; }
    else           { s = e - EREAL; d = s; }          // self loops appended
}

__global__ void edge_max_kernel(const int* __restrict__ src, const int* __restrict__ dst,
                                const float* __restrict__ al_s,
                                const float* __restrict__ al_d,
                                float* __restrict__ m, int hsh)
{
    const unsigned tot = (unsigned)EEXT << hsh;
    const unsigned hmask = (1u << hsh) - 1;
    const unsigned stride = gridDim.x * blockDim.x;
    for (unsigned idx = blockIdx.x * blockDim.x + threadIdx.x; idx < tot; idx += stride) {
        int e = (int)(idx >> hsh), hd = (int)(idx & hmask);
        int s, d; edge_sd(e, src, dst, s, d);
        float v = al_s[(s << hsh) + hd] + al_d[(d << hsh) + hd];
        v = (v > 0.f) ? v : NEGSLOPE * v;             // leaky relu
        atomicMaxF(&m[(d << hsh) + hd], v);
    }
}

__global__ void edge_exp_kernel(const int* __restrict__ src, const int* __restrict__ dst,
                                const float* __restrict__ al_s,
                                const float* __restrict__ al_d,
                                const float* __restrict__ m,
                                float* __restrict__ exb, float* __restrict__ denom, int hsh)
{
    const unsigned tot = (unsigned)EEXT << hsh;
    const unsigned hmask = (1u << hsh) - 1;
    const unsigned stride = gridDim.x * blockDim.x;
    for (unsigned idx = blockIdx.x * blockDim.x + threadIdx.x; idx < tot; idx += stride) {
        int e = (int)(idx >> hsh), hd = (int)(idx & hmask);
        int s, d; edge_sd(e, src, dst, s, d);
        float v = al_s[(s << hsh) + hd] + al_d[(d << hsh) + hd];
        v = (v > 0.f) ? v : NEGSLOPE * v;
        // every segment has a self loop -> m finite (matches isfinite guard)
        float xx = __expf(v - m[(d << hsh) + hd]);
        exb[idx] = xx;
        atomicAdd(&denom[(d << hsh) + hd], xx);
    }
}

__global__ void bias_init_kernel(float* __restrict__ out, const float* __restrict__ b, int fsh)
{
    const unsigned tot = (unsigned)NNODES << fsh;
    const unsigned fmask = (1u << fsh) - 1;
    const unsigned stride = gridDim.x * blockDim.x;
    for (unsigned idx = blockIdx.x * blockDim.x + threadIdx.x; idx < tot; idx += stride)
        out[idx] = b[idx & fmask];
}

// out[dst] += alpha * h[src]  — float4 gathers, per-element atomics.
__global__ void aggregate_kernel(const int* __restrict__ src, const int* __restrict__ dst,
                                 const float* __restrict__ h,
                                 const float* __restrict__ exb,
                                 const float* __restrict__ denom,
                                 float* __restrict__ out,
                                 int fsh, int csh, int hsh)   // Fout=1<<fsh, Cc=1<<csh
{
    const int f4sh = fsh - 2;                      // float4 groups per row
    const unsigned tot = (unsigned)EEXT << f4sh;
    const unsigned qmask = (1u << f4sh) - 1;
    const unsigned stride = gridDim.x * blockDim.x;
    for (unsigned idx = blockIdx.x * blockDim.x + threadIdx.x; idx < tot; idx += stride) {
        int e = (int)(idx >> f4sh);
        int f = (int)(idx & qmask) << 2;
        int hd = f >> csh;
        int s, d; edge_sd(e, src, dst, s, d);
        float alpha = exb[(e << hsh) + hd] / (denom[(d << hsh) + hd] + 1e-16f);
        float4 hv = *(const float4*)(h + ((size_t)s << fsh) + f);
        float* op = out + ((size_t)d << fsh) + f;
        atomicAdd(op + 0, alpha * hv.x);
        atomicAdd(op + 1, alpha * hv.y);
        atomicAdd(op + 2, alpha * hv.z);
        atomicAdd(op + 3, alpha * hv.w);
    }
}

__global__ void elu_kernel(float* __restrict__ x, unsigned n)
{
    const unsigned stride = gridDim.x * blockDim.x;
    for (unsigned i = blockIdx.x * blockDim.x + threadIdx.x; i < n; i += stride) {
        float v = x[i];
        x[i] = (v > 0.f) ? v : (__expf(v) - 1.f);
    }
}

// ----------------------------- orchestration --------------------------------
static inline int ilog2(int v) { int s = 0; while ((1 << s) < v) ++s; return s; }

static void gat_layer(const float* A, int Fin, int Fout, int Hh,
                      const float* W, const float* as_, const float* ad_, const float* b,
                      const int* src, const int* dst,
                      float* hbuf, float* out,
                      float* als, float* ald, float* mbuf, float* dbuf, float* exb,
                      hipStream_t stream)
{
    const int fsh = ilog2(Fout);
    const int hsh = ilog2(Hh);
    const int csh = fsh - hsh;               // Cc = Fout / Hh

    dim3 gg((NNODES + 127) / 128, Fout / 32);
    wmma_gemm_bf16<<<gg, 256, 0, stream>>>(A, W, hbuf, NNODES, Fout, Fin);

    int nh = NNODES * Hh;
    attn_coef_kernel<<<(nh + 255) / 256, 256, 0, stream>>>(hbuf, as_, ad_, als, ald, hsh, csh);
    init_md_kernel<<<(nh + 255) / 256, 256, 0, stream>>>(mbuf, dbuf, nh);

    long long eh = (long long)EEXT * Hh;
    int gb = (int)((eh + 255) / 256); if (gb > 32768) gb = 32768;
    edge_max_kernel<<<gb, 256, 0, stream>>>(src, dst, als, ald, mbuf, hsh);
    edge_exp_kernel<<<gb, 256, 0, stream>>>(src, dst, als, ald, mbuf, exb, dbuf, hsh);

    bias_init_kernel<<<4096, 256, 0, stream>>>(out, b, fsh);
    aggregate_kernel<<<8192, 256, 0, stream>>>(src, dst, hbuf, exb, dbuf, out, fsh, csh, hsh);
}

extern "C" void kernel_launch(void* const* d_in, const int* in_sizes, int n_in,
                              void* d_out, int out_size, void* d_ws, size_t ws_size,
                              hipStream_t stream)
{
    (void)in_sizes; (void)n_in; (void)out_size; (void)ws_size;

    const float* x   = (const float*)d_in[0];
    const int*   ei  = (const int*)d_in[1];       // [2, EREAL]
    const int*   src = ei;
    const int*   dst = ei + EREAL;
    const float* W1  = (const float*)d_in[2];
    const float* as1 = (const float*)d_in[3];
    const float* ad1 = (const float*)d_in[4];
    const float* b1  = (const float*)d_in[5];
    const float* W2  = (const float*)d_in[6];
    const float* as2 = (const float*)d_in[7];
    const float* ad2 = (const float*)d_in[8];
    const float* b2  = (const float*)d_in[9];
    const float* W3  = (const float*)d_in[10];
    const float* as3 = (const float*)d_in[11];
    const float* ad3 = (const float*)d_in[12];
    const float* b3  = (const float*)d_in[13];

    // Workspace carve-up
    char* w = (char*)d_ws;
    auto carve = [&](size_t bytes) {
        float* p = (float*)w;
        w += (bytes + 255) & ~(size_t)255;
        return p;
    };
    float* hbuf = carve((size_t)NNODES * 256 * sizeof(float));  // transformed feats
    float* p1   = carve((size_t)NNODES * 256 * sizeof(float));  // layer1 out
    float* p2   = carve((size_t)NNODES * 256 * sizeof(float));  // layer2 out
    float* als  = carve((size_t)NNODES * 4 * sizeof(float));
    float* ald  = carve((size_t)NNODES * 4 * sizeof(float));
    float* mbuf = carve((size_t)NNODES * 4 * sizeof(float));
    float* dbuf = carve((size_t)NNODES * 4 * sizeof(float));
    float* exb  = carve((size_t)EEXT * 4 * sizeof(float));

    // Layer 1: 256 -> 256, H=4
    gat_layer(x,  256, 256, 4, W1, as1, ad1, b1, src, dst,
              hbuf, p1, als, ald, mbuf, dbuf, exb, stream);
    elu_kernel<<<4096, 256, 0, stream>>>(p1, (unsigned)NNODES * 256);

    // Layer 2: 256 -> 256, H=4
    gat_layer(p1, 256, 256, 4, W2, as2, ad2, b2, src, dst,
              hbuf, p2, als, ald, mbuf, dbuf, exb, stream);
    elu_kernel<<<4096, 256, 0, stream>>>(p2, (unsigned)NNODES * 256);

    // Layer 3: 256 -> 128, H=1 -> d_out
    gat_layer(p2, 256, 128, 1, W3, as3, ad3, b3, src, dst,
              hbuf, (float*)d_out, als, ald, mbuf, dbuf, exb, stream);
}
